// NeuralObliviousDecisionEnsemble_25701084299567
// MI455X (gfx1250) — matrix operand order repaired and checked by reading, not compiled, not run on hardware
//
#include <hip/hip_runtime.h>

typedef __attribute__((ext_vector_type(16))) _Float16 v16h;
typedef __attribute__((ext_vector_type(8)))  float    v8f;
typedef __attribute__((ext_vector_type(4)))  int      v4i;

#define NUM_TREES    64
#define TREE_DEPTH   6
#define NUM_LEAVES   64
#define OUTPUT_DIM   8
#define BATCH        8192
#define NUM_FEATURES 128
#define TD           (NUM_TREES * TREE_DEPTH)   // 384 decision columns

#define WAVES_PER_BLOCK 2
#define ROWS_PER_WAVE   16
#define ROWS_PER_BLOCK  (WAVES_PER_BLOCK * ROWS_PER_WAVE)  // 32
#define BLOCK_THREADS   (WAVES_PER_BLOCK * 32)

// Pre-shuffled f16 B-fragment regions inside d_ws (in _Float16 units):
//   region 0: GEMM1 weights  [nt(24)][kt(4)][lane(32)][e(16)]
//   region 1: GEMM2 leaves   [t(64)][h(2)][lane(32)][e(16)]
#define WB_HALVES (24 * 4 * 32 * 16)   // 49152 halves = 96 KB
#define LB_HALVES (64 * 2 * 32 * 16)   // 65536 halves = 128 KB
#define WS_HALVES (WB_HALVES + LB_HALVES)

// Typed address-space casts for the async Global->LDS builtin
// (param 0 is v4i in AS1 per the compiler diagnostic; LDS side in AS3).
#define GPTR(p) ((__attribute__((address_space(1))) v4i*)(p))
#define LPTR(p) ((__attribute__((address_space(3))) v4i*)(p))

// Fast sigmoid: hardware v_rcp_f32 (~1 ulp) instead of IEEE division.
__device__ __forceinline__ float fast_sigmoid(float x) {
#if __has_builtin(__builtin_amdgcn_rcpf)
    return __builtin_amdgcn_rcpf(1.0f + __expf(-x));
#else
    return 1.0f / (1.0f + __expf(-x));
#endif
}

// ---- WMMA k-order maps (this code's contiguous order, shared by A and B; wave32) ----
// A-matrix 16x32 f16: lanes 0-15 row m, elements 0-15 -> K 0-15;
// lanes 16-31 same rows, K 16-31 (matches kmapB).
__device__ __forceinline__ int kmapA(int e, int lane) {
    return e + 16 * (lane >> 4);
}
// B-matrix 32x16 f16: lane gives N (=lane&15); lanes 0-15 hold K 0-15,
// lanes 16-31 hold K 16-31.
__device__ __forceinline__ int kmapB(int e, int lane) {
    return e + 16 * (lane >> 4);
}

// ---- One-shot prep: convert + shuffle constant operands into WMMA B-fragment
// layout so the hot kernel loads them as two coalesced b128s per fragment. ----
__global__ void prep_fragments_kernel(const float* __restrict__ dweights,
                                      const float* __restrict__ leaves,
                                      _Float16* __restrict__ ws)
{
    const int idx = blockIdx.x * blockDim.x + threadIdx.x;
    if (idx < WB_HALVES) {
        const int e    = idx & 15;
        const int lane = (idx >> 4) & 31;
        const int kt   = (idx >> 9) & 3;
        const int nt   = idx >> 11;
        const int f = kt * 32 + kmapB(e, lane);
        const int c = nt * 16 + (lane & 15);
        ws[idx] = (_Float16)dweights[(size_t)c * NUM_FEATURES + f];
    } else if (idx < WS_HALVES) {
        const int j    = idx - WB_HALVES;
        const int e    = j & 15;
        const int lane = (j >> 4) & 31;
        const int h    = (j >> 9) & 1;
        const int t    = j >> 10;
        const int lb = 32 * h + kmapB(e, lane);
        const int n  = lane & 15;
        ws[idx] = (n < OUTPUT_DIM)
                      ? (_Float16)leaves[((size_t)t * NUM_LEAVES + lb) * OUTPUT_DIM + n]
                      : (_Float16)0.0f;
    }
}

template <bool USE_PRE>
__global__ __launch_bounds__(BLOCK_THREADS)
void node_ensemble_kernel(const float* __restrict__ inputs,     // (8192,128)
                          const float* __restrict__ dweights,   // (64,6,128)
                          const float* __restrict__ dbiases,    // (64,6)
                          const float* __restrict__ leaves,     // (64,64,8)
                          const _Float16* __restrict__ pre,     // d_ws fragments
                          float* __restrict__ out)              // (8192,8)
{
    __shared__ float    s_in[ROWS_PER_BLOCK][NUM_FEATURES];   // 16 KB
    __shared__ _Float16 s_dec[ROWS_PER_BLOCK][TD];            // 24 KB

    const int lane = threadIdx.x & 31;
    const int wave = threadIdx.x >> 5;
    const int m    = lane & 15;          // row within tile / N column
    const int hi   = lane >> 4;          // half-wave select
    const int blockRow = blockIdx.x * ROWS_PER_BLOCK;
    const int waveRow  = blockRow + wave * ROWS_PER_WAVE;
    const int rw       = wave * ROWS_PER_WAVE + m;  // this lane's LDS row

    // ---- Stage the input tile to LDS; prefer the async Global->LDS path ----
#if __has_builtin(__builtin_amdgcn_global_load_async_to_lds_b128)
    {
        const float* srcBase = inputs + (size_t)blockRow * NUM_FEATURES;
        for (int i = threadIdx.x; i < ROWS_PER_BLOCK * NUM_FEATURES / 4; i += BLOCK_THREADS) {
            const float* src = srcBase + i * 4;
            float* dst = &s_in[0][0] + i * 4;
            __builtin_amdgcn_global_load_async_to_lds_b128(GPTR(src), LPTR(dst), 0, 0);
        }
#if __has_builtin(__builtin_amdgcn_s_wait_asynccnt)
        __builtin_amdgcn_s_wait_asynccnt(0);
#else
        asm volatile("s_wait_asynccnt 0x0" ::: "memory");
#endif
    }
#else
    for (int i = threadIdx.x; i < ROWS_PER_BLOCK * NUM_FEATURES / 4; i += BLOCK_THREADS) {
        const float4 v = reinterpret_cast<const float4*>(
            inputs + (size_t)blockRow * NUM_FEATURES)[i];
        reinterpret_cast<float4*>(&s_in[0][0])[i] = v;
    }
#endif
    __syncthreads();

    // ---- Build persistent A fragments for GEMM1: 4 k-tiles of 16x32 f16 ----
    v16h afrag[4];
#pragma unroll
    for (int kt = 0; kt < 4; ++kt) {
#pragma unroll
        for (int e = 0; e < 16; ++e) {
            const int f = kt * 32 + kmapA(e, lane);
            afrag[kt][e] = (_Float16)s_in[rw][f];
        }
    }

    // ---- GEMM1: logits (16 x 384, K=128) -> sigmoid -> s_dec (f16) ----
    for (int nt = 0; nt < TD / 16; ++nt) {          // 24 N-tiles
        const int c = nt * 16 + m;                  // decision column = t*6+d
        v8f cf = {};
#pragma unroll
        for (int kt = 0; kt < 4; ++kt) {
            v16h bfrag;
            if constexpr (USE_PRE) {
                bfrag = reinterpret_cast<const v16h*>(pre)[(nt * 4 + kt) * 32 + lane];
            } else {
#pragma unroll
                for (int e = 0; e < 16; ++e) {
                    const int f = kt * 32 + kmapB(e, lane);
                    bfrag[e] = (_Float16)dweights[(size_t)c * NUM_FEATURES + f];
                }
            }
            cf = __builtin_amdgcn_wmma_f32_16x16x32_f16(
                     false, afrag[kt], false, bfrag, (short)0, cf, false, false);
        }
        const float bias = dbiases[c];
#pragma unroll
        for (int r = 0; r < 8; ++r) {
            const int mr = r + 8 * hi;              // row of this accumulator elem
            const float s = fast_sigmoid(cf[r] - bias);
            s_dec[wave * ROWS_PER_WAVE + mr][c] = (_Float16)s;
        }
    }
    __syncthreads();

    // ---- GEMM2: out = (1/64) * P(16 x 4096) @ L(4096 x 16-padded) ----
    // K chunked as (tree, leaf-half): 64 trees * 2 halves = 128 WMMA.
    v8f acc = {};
    for (int t = 0; t < NUM_TREES; ++t) {
        const float p0 = (float)s_dec[rw][t * TREE_DEPTH + 0];
        const float p1 = (float)s_dec[rw][t * TREE_DEPTH + 1];
        const float p2 = (float)s_dec[rw][t * TREE_DEPTH + 2];
        const float p3 = (float)s_dec[rw][t * TREE_DEPTH + 3];
        const float p4 = (float)s_dec[rw][t * TREE_DEPTH + 4];
        const float p5 = (float)s_dec[rw][t * TREE_DEPTH + 5];
        float lo[8], hv[8];
#pragma unroll
        for (int i = 0; i < 8; ++i) {
            lo[i] = ((i & 1) ? p0 : 1.0f - p0) *
                    ((i & 2) ? p1 : 1.0f - p1) *
                    ((i & 4) ? p2 : 1.0f - p2);
            hv[i] = ((i & 1) ? p3 : 1.0f - p3) *
                    ((i & 2) ? p4 : 1.0f - p4) *
                    ((i & 4) ? p5 : 1.0f - p5);
        }
#pragma unroll
        for (int h = 0; h < 2; ++h) {               // leaves 0-31, 32-63
            v16h aF, bF;
#pragma unroll
            for (int e = 0; e < 16; ++e) {
                const int ka = kmapA(e, lane);      // leaf index within half
                aF[e] = (_Float16)(lo[ka & 7] * hv[(ka >> 3) + 4 * h]);
            }
            if constexpr (USE_PRE) {
                bF = reinterpret_cast<const v16h*>(pre + WB_HALVES)[(t * 2 + h) * 32 + lane];
            } else {
#pragma unroll
                for (int e = 0; e < 16; ++e) {
                    const int lb = 32 * h + kmapB(e, lane);
                    const float bv = (m < OUTPUT_DIM)
                        ? leaves[((size_t)t * NUM_LEAVES + lb) * OUTPUT_DIM + m]
                        : 0.0f;
                    bF[e] = (_Float16)bv;
                }
            }
            acc = __builtin_amdgcn_wmma_f32_16x16x32_f16(
                      false, aF, false, bF, (short)0, acc, false, false);
        }
    }

    // ---- Write back: mean over trees = acc / 64 ----
    if (m < OUTPUT_DIM) {
#pragma unroll
        for (int r = 0; r < 8; ++r) {
            const int mr = r + 8 * hi;
            out[(size_t)(waveRow + mr) * OUTPUT_DIM + m] = acc[r] * (1.0f / 64.0f);
        }
    }
}

extern "C" void kernel_launch(void* const* d_in, const int* in_sizes, int n_in,
                              void* d_out, int out_size, void* d_ws, size_t ws_size,
                              hipStream_t stream) {
    const float* inputs   = (const float*)d_in[0];
    const float* dweights = (const float*)d_in[1];
    const float* dbiases  = (const float*)d_in[2];
    const float* leaves   = (const float*)d_in[3];
    float* out = (float*)d_out;

    const dim3 grid(BATCH / ROWS_PER_BLOCK);   // 256 blocks
    const dim3 block(BLOCK_THREADS);           // 64 threads = 2 waves

    const bool use_pre = ws_size >= (size_t)WS_HALVES * sizeof(_Float16);
    if (use_pre) {
        _Float16* ws = (_Float16*)d_ws;
        prep_fragments_kernel<<<(WS_HALVES + 255) / 256, 256, 0, stream>>>(
            dweights, leaves, ws);
        node_ensemble_kernel<true><<<grid, block, 0, stream>>>(
            inputs, dweights, dbiases, leaves, ws, out);
    } else {
        node_ensemble_kernel<false><<<grid, block, 0, stream>>>(
            inputs, dweights, dbiases, leaves, nullptr, out);
    }
    (void)in_sizes; (void)n_in; (void)out_size;
}
